// CNN_4870492914291
// MI455X (gfx1250) — matrix operand order repaired and checked by reading, not compiled, not run on hardware
//
#include <hip/hip_runtime.h>
#include <hip/hip_bf16.h>

// CDNA5 (gfx1250) wave32 WMMA implementation of the 10-layer non-overlapping
// conv chain. Each layer: GEMM M=256, K=512 (cin*P), N=128*q on
// V_WMMA_F32_16X16X32_F16 (f32 accumulate).
//
// Layout trick: activations stored transposed [b][x][c] and weight rows
// reordered to j = k*256 + c, so every B-fragment address is linear in the
// WMMA contraction index j -> each lane's fragment is 32 contiguous bytes
// (2x global_load_b128), and epilogue stores are packed b128s.

typedef __attribute__((ext_vector_type(16))) _Float16     v16h;
typedef __attribute__((ext_vector_type(8)))  float        v8f;
typedef __attribute__((ext_vector_type(2)))  _Float16     h2;
typedef __attribute__((ext_vector_type(2)))  float        f2;
typedef __attribute__((ext_vector_type(4)))  unsigned int v4u;
typedef __attribute__((ext_vector_type(8)))  unsigned int v8u;

constexpr int ilog2c(int v) { int r = 0; while (v > 1) { v >>= 1; ++r; } return r; }

// ---- fragment loaders -------------------------------------------------------
// A 16x32 f16 fragment, contiguous k order used by this code: lanes 0-15 hold K 0..15, lanes 16-31 K 16..31;
// with row-major [o][512] both K groups are 4 contiguous dwords.
// p must already include the +16*hi half-offset.
__device__ __forceinline__ v16h load_afrag(const _Float16* __restrict__ p) {
    v4u g0 = *(const v4u*)p;          // K = k0   .. k0+7   (+16*hi)
    v4u g1 = *(const v4u*)(p + 8);    // K = k0+8 .. k0+15  (+16*hi)
    v8u u;
    u[0] = g0[0]; u[1] = g0[1]; u[2] = g0[2]; u[3] = g0[3];
    u[4] = g1[0]; u[5] = g1[1]; u[6] = g1[2]; u[7] = g1[3];
    return __builtin_bit_cast(v16h, u);
}
// B 32x16 f16 fragment, same contiguous k order as A: lanes 0-15 K 0-15, lanes 16-31 K 16-31; with the
// linear-j layout this is one 32-byte contiguous run (p includes +16*hi).
__device__ __forceinline__ v16h load_bfrag(const _Float16* __restrict__ p) {
    return __builtin_bit_cast(v16h, *(const v8u*)p);
}

// ---- epilogue stores: scale (1/sqrt(256)) + relu ---------------------------
__device__ __forceinline__ void store_tile_f16(const v8f& acc, _Float16* __restrict__ p) {
    v4u u;
#pragma unroll
    for (int r = 0; r < 4; ++r) {
        h2 hh;
        hh[0] = (_Float16)fmaxf(acc[2 * r]     * 0.0625f, 0.0f);
        hh[1] = (_Float16)fmaxf(acc[2 * r + 1] * 0.0625f, 0.0f);
        u[r] = __builtin_bit_cast(unsigned int, hh);
    }
    *(v4u*)p = u;                       // 16B packed store
}
__device__ __forceinline__ void store_tile_f32(const v8f& acc, float* __restrict__ p) {
    v8f o;
#pragma unroll
    for (int r = 0; r < 8; ++r) o[r] = fmaxf(acc[r] * 0.0625f, 0.0f);
    *(v8f*)p = o;                       // 32B packed store
}

// ---- weight prep: f32 -> f16, reorder rows to j=k*256+c for layers 1..9 ----
__global__ __launch_bounds__(256) void weight_prep_kernel(
    const float* __restrict__ src, _Float16* __restrict__ dst) {
    const int t   = blockIdx.x * blockDim.x + threadIdx.x;   // 10*256*512
    const int row = t >> 9;                                  // l*256 + o
    const int jj  = t & 511;
    const int l   = row >> 8;
    int srcIdx;
    if (l == 0) srcIdx = row * 512 + jj;                     // keep c-major
    else { const int k = jj >> 8, c = jj & 255; srcIdx = row * 512 + c * 2 + k; }
    dst[t] = (_Float16)src[srcIdx];
}

// ---- layer 0: fp32 x [128][256][1024] -> f16 out [128][512][256] -----------
__global__ __launch_bounds__(128) void conv0_kernel(
    const float* __restrict__ x, const _Float16* __restrict__ w,
    _Float16* __restrict__ out) {
    constexpr int D = 1024, Q = 512, QBITS = 9;
    const int lane = threadIdx.x & 31;
    const int wave = threadIdx.x >> 5;
    const int lo = lane & 15, hi = lane >> 4;
    const int tile = blockIdx.x * 4 + wave;
    const int m0 = (tile & 15) * 16;
    const int n0 = (tile >> 4) * 64;

    const _Float16* __restrict__ pA = w + (m0 + lo) * 512 + 16 * hi;

    const float* xb[4]; int bc[4], qc[4];
#pragma unroll
    for (int t = 0; t < 4; ++t) {
        const int n = n0 + 16 * t + lo;
        bc[t] = n >> QBITS; qc[t] = n & (Q - 1);
        xb[t] = x + bc[t] * (256 * D) + 2 * qc[t];
    }

    v8f acc[4] = {};
    for (int k0 = 0; k0 < 512; k0 += 32) {
        const v16h a = load_afrag(pA + k0);
        const int c0 = k0 >> 1;
#pragma unroll
        for (int t = 0; t < 4; ++t) {
            v8u u;
#pragma unroll
            for (int i = 0; i < 8; ++i) {
                const int c = c0 + i + 8 * hi;               // c-major j for layer 0
                f2 f = *(const f2*)(xb[t] + c * D);          // k=0,1 contiguous
                h2 hh; hh[0] = (_Float16)f[0]; hh[1] = (_Float16)f[1];
                u[i] = __builtin_bit_cast(unsigned int, hh);
            }
            acc[t] = __builtin_amdgcn_wmma_f32_16x16x32_f16(
                false, a, false, __builtin_bit_cast(v16h, u), (short)0, acc[t], false, false);
        }
    }
#pragma unroll
    for (int t = 0; t < 4; ++t)
        store_tile_f16(acc[t], out + (bc[t] * Q + qc[t]) * 256 + m0 + 8 * hi);
}

// ---- layers 1..9: f16 in [128][D][256] -> f16 out [128][D/2][256] ----------
template<int D>
__global__ __launch_bounds__(128) void convB_kernel(
    const _Float16* __restrict__ in,
    const _Float16* __restrict__ w,       // rows in j=k*256+c order
    _Float16* __restrict__ out,
    float* __restrict__ out_f32) {        // layer 9 only: y_hier [128][256]
    constexpr int Q = D / 2;
    constexpr int QBITS = ilog2c(Q);
    const int lane = threadIdx.x & 31;
    const int wave = threadIdx.x >> 5;
    const int lo = lane & 15, hi = lane >> 4;
    const int tile = blockIdx.x * 4 + wave;   // tiles = 16*(128*Q/64), mult of 4
    const int m0 = (tile & 15) * 16;
    const int n0 = (tile >> 4) * 64;

    const _Float16* __restrict__ pA = w + (m0 + lo) * 512 + 16 * hi;

    const _Float16* pB[4]; int bc[4], qc[4];
#pragma unroll
    for (int t = 0; t < 4; ++t) {
        const int n = n0 + 16 * t + lo;
        bc[t] = n >> QBITS; qc[t] = n & (Q - 1);
        // address of j=0 for column (b,q): in[(b*D + 2q)*256 + j]
        pB[t] = in + (bc[t] * D + 2 * qc[t]) * 256 + 16 * hi;
    }

    v8f acc[4] = {};
#pragma unroll 4
    for (int k0 = 0; k0 < 512; k0 += 32) {
        const v16h a = load_afrag(pA + k0);
#pragma unroll
        for (int t = 0; t < 4; ++t) {
            const v16h b = load_bfrag(pB[t] + k0);           // 32B contiguous
            acc[t] = __builtin_amdgcn_wmma_f32_16x16x32_f16(
                false, a, false, b, (short)0, acc[t], false, false);
        }
    }

#pragma unroll
    for (int t = 0; t < 4; ++t) {
        store_tile_f16(acc[t], out + (bc[t] * Q + qc[t]) * 256 + m0 + 8 * hi);
        if (out_f32)                                          // Q==1 here
            store_tile_f32(acc[t], out_f32 + bc[t] * 256 + m0 + 8 * hi);
    }
}

// ---- final: y[b,m] = (1/256) * sum_h y_hier[b,h] * beta[h,m] ---------------
__global__ __launch_bounds__(256) void final_gemm_kernel(
    const float* __restrict__ yh, const float* __restrict__ beta,
    float* __restrict__ y) {
    const int t = blockIdx.x * blockDim.x + threadIdx.x;     // 16384 threads
    const int b = t >> 7, m = t & 127;
    float s = 0.0f;
    for (int h = 0; h < 256; ++h) s += yh[b * 256 + h] * beta[h * 128 + m];
    y[t] = s * (1.0f / 256.0f);
}

// ---- host-side sequencing ---------------------------------------------------
template<int D>
static inline void run_layer(const _Float16* in, const _Float16* w,
                             _Float16* out, float* of32, hipStream_t s) {
    constexpr int Q = D / 2;
    constexpr int blocks = (16 * (128 * Q / 64)) / 4;        // = 8*Q, exact
    convB_kernel<D><<<blocks, 128, 0, s>>>(in, w, out, of32);
}

extern "C" void kernel_launch(void* const* d_in, const int* in_sizes, int n_in,
                              void* d_out, int out_size, void* d_ws, size_t ws_size,
                              hipStream_t stream) {
    (void)in_sizes; (void)n_in; (void)out_size; (void)ws_size;

    const float* x    = (const float*)d_in[0];   // [128][256][1024]
    const float* wsrc = (const float*)d_in[1];   // [10][256][256][2]
    const float* beta = (const float*)d_in[2];   // [256][128]

    float* y     = (float*)d_out;                // [128][128]
    float* yhier = (float*)d_out + 128 * 128;    // [128][256][1]

    // Workspace (~52 MB):
    //   [0, 2.6MB)   f16 weights (layer-0 c-major, layers 1-9 k-major rows)
    //   [4MB, 36MB)  bufA (f16), up to 128*512*256
    //   [36MB, 52MB) bufB (f16), up to 128*256*256
    char* wsb = (char*)d_ws;
    _Float16* wHalf = (_Float16*)wsb;
    _Float16* bufA  = (_Float16*)(wsb + (size_t(4)  << 20));
    _Float16* bufB  = (_Float16*)(wsb + (size_t(36) << 20));

    weight_prep_kernel<<<(10 * 256 * 512) / 256, 256, 0, stream>>>(wsrc, wHalf);

    // Layer 0: 8*Q = 4096 blocks.
    conv0_kernel<<<8 * 512, 128, 0, stream>>>(x, wHalf, bufA);

    constexpr int WROW = 256 * 512;
    run_layer<512>(bufA, wHalf + 1 * WROW, bufB, nullptr, stream);
    run_layer<256>(bufB, wHalf + 2 * WROW, bufA, nullptr, stream);
    run_layer<128>(bufA, wHalf + 3 * WROW, bufB, nullptr, stream);
    run_layer< 64>(bufB, wHalf + 4 * WROW, bufA, nullptr, stream);
    run_layer< 32>(bufA, wHalf + 5 * WROW, bufB, nullptr, stream);
    run_layer< 16>(bufB, wHalf + 6 * WROW, bufA, nullptr, stream);
    run_layer<  8>(bufA, wHalf + 7 * WROW, bufB, nullptr, stream);
    run_layer<  4>(bufB, wHalf + 8 * WROW, bufA, nullptr, stream);
    run_layer<  2>(bufA, wHalf + 9 * WROW, bufB, yhier, stream);

    final_gemm_kernel<<<64, 256, 0, stream>>>(yhier, beta, y);
}
